// MSMDIA_8186207666343
// MI455X (gfx1250) — matrix operand order repaired and checked by reading, not compiled, not run on hardware
//
#include <hip/hip_runtime.h>
#include <hip/hip_bf16.h>

typedef _Float16 half_t;
typedef __attribute__((ext_vector_type(16))) _Float16 v16h;
typedef __attribute__((ext_vector_type(8)))  _Float16 h8;
typedef __attribute__((ext_vector_type(8)))  float    v8f;

constexpr int CB = 2;          // batch
constexpr int CC = 48;         // channels
constexpr int CH_ = 128;       // H
constexpr int CW_ = 256;       // W
constexpr int CP = CH_ * CW_;  // 32768 pixels
constexpr int CBP = CB * CP;   // 65536

// ---------------- fragment helpers (CDNA5 WMMA f16 16x16x32 layouts) ----------

union F16x16 { v16h v; h8 h[2]; };

// A (16x32, MxK): lane m+16h -> elems[0..7]=A[m][kb+8h..], elems[8..15]=A[m][kb+16+8h..]
__device__ __forceinline__ v16h load_a_frag(const half_t* row, int kb, int h) {
  F16x16 u;
  u.h[0] = *(const h8*)(row + kb + 8 * h);
  u.h[1] = *(const h8*)(row + kb + 16 + 8 * h);
  return u.v;
}
// B (32x16, KxN): lane n+16h -> elems[0..7]=B[kb+8h..][n], elems[8..15]=B[kb+16+8h..][n]
__device__ __forceinline__ v16h load_b_frag(const half_t* row, int kb, int h) {
  F16x16 u;
  u.h[0] = *(const h8*)(row + kb + 8 * h);
  u.h[1] = *(const h8*)(row + kb + 16 + 8 * h);
  return u.v;
}
__device__ __forceinline__ v8f wmma_f16(v16h a, v16h b, v8f c) {
  return __builtin_amdgcn_wmma_f32_16x16x32_f16(false, a, false, b, (short)0, c,
                                                false, false);
}

// ---------------- K0: weight prepack to f16 ----------------------------------
// wqkv: [144][64] (K zero-padded 48->64), wdil: 3x [96][tap*96+ic], wq: [48][288]
__global__ void k_prepack(const float* qkvLw, const float* qkvRw, const float* d0,
                          const float* d1, const float* d2, const float* qw,
                          half_t* wqkvL, half_t* wqkvR, half_t* wdil, half_t* wq) {
  int t = blockIdx.x * 256 + threadIdx.x;
  const int N1 = 144 * 64;
  if (t < 2 * N1) {
    int side = t / N1, r = t % N1;
    int o = r >> 6, k = r & 63;
    const float* src = side ? qkvRw : qkvLw;
    (side ? wqkvR : wqkvL)[r] = (half_t)((k < 48) ? src[o * 48 + k] : 0.0f);
    return;
  }
  t -= 2 * N1;
  const int N2 = 96 * 864;
  if (t < 3 * N2) {
    int d = t / N2, r = t % N2;
    int o = r / 864, k = r % 864;
    int tap = k / 96, i = k % 96;
    const float* src = (d == 0) ? d0 : ((d == 1) ? d1 : d2);
    wdil[d * N2 + r] = (half_t)src[(o * 96 + i) * 9 + tap];  // OIHW 3x3
    return;
  }
  t -= 3 * N2;
  if (t < 48 * 288) wq[t] = (half_t)qw[t];
}

// ---------------- K1: LayerNorm2d -> f16 pixel-major [B][P][48] --------------
__global__ void k_ln(const float* xl, const float* xr, const float* w1,
                     const float* b1, const float* w2, const float* b2,
                     half_t* normL, half_t* normR) {
  int t = blockIdx.x * 256 + threadIdx.x;  // 2*B*P threads
  int p = t & (CP - 1); t >>= 15;
  int b = t & 1, side = t >> 1;
  const float* x = side ? xr : xl;
  const float* w = side ? w2 : w1;
  const float* bb = side ? b2 : b1;
  half_t* out = (side ? normR : normL) + (size_t)(b * CP + p) * 48;
  float v[48], mu = 0.f;
#pragma unroll
  for (int c = 0; c < 48; c++) { v[c] = x[(size_t)(b * 48 + c) * CP + p]; mu += v[c]; }
  mu *= (1.0f / 48.0f);
  float var = 0.f;
#pragma unroll
  for (int c = 0; c < 48; c++) { float d = v[c] - mu; var += d * d; }
  float rs = rsqrtf(var * (1.0f / 48.0f) + 1e-6f);
#pragma unroll
  for (int c = 0; c < 48; c++) out[c] = (half_t)((v[c] - mu) * rs * w[c] + bb[c]);
}

// ---------------- K2: qkv 1x1 (48->144) WMMA GEMM, pixel-major f16 out -------
__global__ void k_qkv(const half_t* normL, const half_t* normR, const half_t* wL,
                      const half_t* wR, const float* bL, const float* bR,
                      half_t* qkvL, half_t* qkvR) {
  int gw = (blockIdx.x * 256 + threadIdx.x) >> 5;
  int lane = threadIdx.x & 31, m = lane & 15, h = lane >> 4;
  int ng = gw & 511; gw >>= 9;
  int mt = gw % 9; gw /= 9;
  int b = gw & 1, side = gw >> 1;
  const half_t* norm = side ? normR : normL;
  const half_t* wt = side ? wR : wL;
  const float* bias = side ? bR : bL;
  half_t* out = side ? qkvR : qkvL;
  const half_t* arow = wt + (mt * 16 + m) * 64;
  v16h a0 = load_a_frag(arow, 0, h);
  v16h a1 = load_a_frag(arow, 32, h);
#pragma unroll
  for (int sub = 0; sub < 4; sub++) {
    int p = ng * 64 + sub * 16 + m;
    const half_t* brow = norm + (size_t)(b * CP + p) * 48;
    v8f acc = {};
    acc = wmma_f16(a0, load_b_frag(brow, 0, h), acc);
    acc = wmma_f16(a1, load_b_frag(brow, 32, h), acc);  // k>=48 hits zero weights
    h8 st;
#pragma unroll
    for (int r = 0; r < 8; r++)
      st[r] = (half_t)(acc[r] + bias[mt * 16 + 8 * h + r]);
    // pixel-major [B][P][144]: 8 consecutive oc per lane -> one b128 store
    *(h8*)(out + (size_t)(b * CP + p) * 144 + mt * 16 + 8 * h) = st;
  }
}

// ---------------- K3: depthwise 3x3 + split/scatter (c-fastest mapping) ------
__global__ void k_dw(const half_t* qkvL, const half_t* qkvR, const float* dwLw,
                     const float* dwLb, const float* dwRw, const float* dwRb,
                     half_t* Qcat, float* KL, float* KR, half_t* VL, half_t* VR) {
  size_t t = (size_t)blockIdx.x * 256 + threadIdx.x;  // 2*B*P*144
  int c = (int)(t % 144); t /= 144;
  int p = (int)(t & (CP - 1)); t >>= 15;
  int b = (int)(t & 1), side = (int)(t >> 1);
  const half_t* in = (side ? qkvR : qkvL) + (size_t)b * CP * 144;
  const float* wp = (side ? dwRw : dwLw) + c * 9;
  float s = (side ? dwRb : dwLb)[c];
  int y = p >> 8, x = p & 255;
#pragma unroll
  for (int ky = 0; ky < 3; ky++) {
    int yy = y + ky - 1;
    if ((unsigned)yy >= (unsigned)CH_) continue;
#pragma unroll
    for (int kx = 0; kx < 3; kx++) {
      int xx = x + kx - 1;
      if ((unsigned)xx >= (unsigned)CW_) continue;
      s += wp[ky * 3 + kx] * (float)in[(size_t)(yy * 256 + xx) * 144 + c];
    }
  }
  if (c < 48)       Qcat[(size_t)(b * CP + p) * 96 + side * 48 + c] = (half_t)s;
  else if (c < 96)  (side ? KR : KL)[(size_t)(b * 48 + (c - 48)) * CP + p] = s;
  else              (side ? VR : VL)[(size_t)(b * CP + p) * 48 + (c - 96)] = (half_t)s;
}

// ---------------- K4: dilated 3x3 convs (96->96, d=2/4/6) WMMA ---------------
__global__ void k_dil(const half_t* Qcat, const half_t* wdil, const float* b0,
                      const float* b1, const float* b2, half_t* dcat) {
  int gw = (blockIdx.x * 256 + threadIdx.x) >> 5;
  int lane = threadIdx.x & 31, m = lane & 15, h = lane >> 4;
  int ng = gw & 511; gw >>= 9;
  int mt = gw % 6; gw /= 6;
  int b = gw & 1, d = gw >> 1;
  int dil = 2 * (d + 1);
  const half_t* wt = wdil + d * (96 * 864) + (mt * 16 + m) * 864;
  const float* bias = (d == 0) ? b0 : ((d == 1) ? b1 : b2);
  int pn = ng * 64 + m;
  // Warm the WGP cache with the center-tap rows (re-read by all 9 taps).
#pragma unroll
  for (int sub = 0; sub < 4; sub++)
    __builtin_prefetch(Qcat + (size_t)(b * CP + pn + sub * 16) * 96, 0, 1);
  v8f acc[4] = {};
#pragma unroll
  for (int tap = 0; tap < 9; tap++) {
    int dy = tap / 3 - 1, dx = tap % 3 - 1;
#pragma unroll
    for (int ks = 0; ks < 3; ks++) {
      v16h af = load_a_frag(wt, tap * 96 + ks * 32, h);
#pragma unroll
      for (int sub = 0; sub < 4; sub++) {
        int p = pn + sub * 16;
        int y = (p >> 8) + dy * dil, x = (p & 255) + dx * dil;
        v16h bf = {};
        if ((unsigned)y < (unsigned)CH_ && (unsigned)x < (unsigned)CW_)
          bf = load_b_frag(Qcat + (size_t)(b * CP + y * 256 + x) * 96, ks * 32, h);
        acc[sub] = wmma_f16(af, bf, acc[sub]);
      }
    }
  }
#pragma unroll
  for (int sub = 0; sub < 4; sub++) {
    int p = pn + sub * 16;
    h8 st;
#pragma unroll
    for (int r = 0; r < 8; r++) st[r] = (half_t)(acc[sub][r] + bias[mt * 16 + 8 * h + r]);
    *(h8*)(dcat + (size_t)(b * CP + p) * 288 + d * 96 + mt * 16 + 8 * h) = st;
  }
}

// ---------------- K5: convQ 1x1 (288->48) WMMA, fp32 out channel-major -------
__global__ void k_convq(const half_t* dcat, const half_t* wq, const float* qb,
                        float* Qfin) {
  int gw = (blockIdx.x * 256 + threadIdx.x) >> 5;
  int lane = threadIdx.x & 31, m = lane & 15, h = lane >> 4;
  int ng = gw & 511; gw >>= 9;
  int mt = gw % 3; gw /= 3;
  int b = gw;
  const half_t* arow = wq + (mt * 16 + m) * 288;
  v8f acc[4] = {};
#pragma unroll
  for (int ks = 0; ks < 9; ks++) {
    v16h af = load_a_frag(arow, ks * 32, h);
#pragma unroll
    for (int sub = 0; sub < 4; sub++) {
      int p = ng * 64 + sub * 16 + m;
      acc[sub] = wmma_f16(af, load_b_frag(dcat + (size_t)(b * CP + p) * 288, ks * 32, h),
                          acc[sub]);
    }
  }
#pragma unroll
  for (int sub = 0; sub < 4; sub++) {
    int p = ng * 64 + sub * 16 + m;
#pragma unroll
    for (int r = 0; r < 8; r++) {
      int oc = mt * 16 + 8 * h + r;
      Qfin[(size_t)(b * 48 + oc) * CP + p] = acc[sub][r] + qb[oc];
    }
  }
}

// ---------------- K6: 1/max(||row||,eps) for Q, K_l, K_r ---------------------
__global__ void k_l2inv(const float* Qfin, const float* KL, const float* KR,
                        float* invQ, float* invKL, float* invKR) {
  __shared__ float red[256];
  int idx = blockIdx.x;
  int c = idx % 48; idx /= 48;
  int b = idx & 1; idx >>= 1;
  const float* src = (idx == 0) ? Qfin : ((idx == 1) ? KL : KR);
  float* dst = (idx == 0) ? invQ : ((idx == 1) ? invKL : invKR);
  const float* row = src + (size_t)(b * 48 + c) * CP;
  float s = 0.f;
  for (int i = threadIdx.x; i < CP; i += 256) { float v = row[i]; s += v * v; }
  red[threadIdx.x] = s; __syncthreads();
  for (int o = 128; o > 0; o >>= 1) {
    if (threadIdx.x < o) red[threadIdx.x] += red[threadIdx.x + o];
    __syncthreads();
  }
  if (threadIdx.x == 0) dst[b * 48 + c] = 1.0f / fmaxf(sqrtf(red[0]), 1e-12f);
}

// ---------------- K7: 16x16 channel-attention scores + softmax ---------------
__global__ void k_attn(const float* Qfin, const float* KL, const float* KR,
                       const float* invQ, const float* invKL, const float* invKR,
                       const float* temp, float* A) {
  __shared__ float Qs[16][128];
  __shared__ float Ks[16][128];
  __shared__ float Sv[256];
  int idx = blockIdx.x;
  int head = idx % 3; idx /= 3;
  int b = idx & 1, side = idx >> 1;             // side0 -> F_l uses K_r
  const float* K = side ? KL : KR;
  const float* invK = side ? invKL : invKR;
  int tid = threadIdx.x, c = tid >> 4, dd = tid & 15;
  const float* Qb = Qfin + (size_t)(b * 48 + head * 16) * CP;
  const float* Kb = K + (size_t)(b * 48 + head * 16) * CP;
  float s = 0.f;
  for (int n0 = 0; n0 < CP; n0 += 128) {
#pragma unroll
    for (int j = 0; j < 8; j++) {
      int li = tid * 8 + j, row = li >> 7, col = li & 127;
      Qs[row][col] = Qb[(size_t)row * CP + n0 + col];
      Ks[row][col] = Kb[(size_t)row * CP + n0 + col];
    }
    __syncthreads();
#pragma unroll 8
    for (int j = 0; j < 128; j++) s += Qs[c][j] * Ks[dd][j];
    __syncthreads();
  }
  s *= invQ[b * 48 + head * 16 + c] * invK[b * 48 + head * 16 + dd] * temp[head];
  Sv[tid] = s; __syncthreads();
  float mx = -1e30f;
#pragma unroll
  for (int j = 0; j < 16; j++) mx = fmaxf(mx, Sv[c * 16 + j]);
  __syncthreads();
  float e = __expf(s - mx);
  Sv[tid] = e; __syncthreads();
  float sum = 0.f;
#pragma unroll
  for (int j = 0; j < 16; j++) sum += Sv[c * 16 + j];
  A[(size_t)((side * 2 + b) * 3 + head) * 256 + tid] = e / sum;
}

// ---------------- K8: fold conv1/conv2 into attention: Wc = W . blockdiag(A) -
__global__ void k_wc(const float* A, const float* w1, const float* w2, half_t* Wc) {
  int t = blockIdx.x * 256 + threadIdx.x;  // 2*2*48*64
  int k = t & 63; t >>= 6;
  int o = t % 48; t /= 48;
  int b = t & 1, side = t >> 1;
  float s = 0.f;
  if (k < 48) {
    int hh = k >> 4, dk = k & 15;
    const float* w = side ? w2 : w1;
    const float* Ab = A + (size_t)((side * 2 + b) * 3 + hh) * 256;
#pragma unroll
    for (int c = 0; c < 16; c++) s += w[o * 48 + hh * 16 + c] * Ab[c * 16 + dk];
  }
  Wc[(size_t)((side * 2 + b) * 48 + o) * 64 + k] = (half_t)s;
}

// ---------------- K9: out = x + Wc . V + bias (WMMA) -------------------------
__global__ void k_out(const half_t* VL, const half_t* VR, const half_t* Wc,
                      const float* xl, const float* xr, const float* b1,
                      const float* b2, float* out) {
  int gw = (blockIdx.x * 256 + threadIdx.x) >> 5;
  int lane = threadIdx.x & 31, m = lane & 15, h = lane >> 4;
  int ng = gw & 511; gw >>= 9;
  int mt = gw % 3; gw /= 3;
  int b = gw & 1, side = gw >> 1;
  const half_t* V = side ? VR : VL;
  const float* x = side ? xr : xl;
  const float* bias = side ? b2 : b1;
  const half_t* arow = Wc + (size_t)((side * 2 + b) * 48 + mt * 16 + m) * 64;
  v16h a0 = load_a_frag(arow, 0, h);
  v16h a1 = load_a_frag(arow, 32, h);
  float* o = out + (size_t)side * ((size_t)CB * 48 * CP);
#pragma unroll
  for (int sub = 0; sub < 4; sub++) {
    int p = ng * 64 + sub * 16 + m;
    const half_t* brow = V + (size_t)(b * CP + p) * 48;
    v8f acc = {};
    acc = wmma_f16(a0, load_b_frag(brow, 0, h), acc);
    acc = wmma_f16(a1, load_b_frag(brow, 32, h), acc);  // k>=48 zero weights
#pragma unroll
    for (int r = 0; r < 8; r++) {
      int oc = mt * 16 + 8 * h + r;
      size_t off = (size_t)(b * 48 + oc) * CP + p;
      o[off] = x[off] + acc[r] + bias[oc];
    }
  }
}

// ---------------- host side --------------------------------------------------
extern "C" void kernel_launch(void* const* d_in, const int* in_sizes, int n_in,
                              void* d_out, int out_size, void* d_ws, size_t ws_size,
                              hipStream_t stream) {
  const float* x_l   = (const float*)d_in[0];
  const float* x_r   = (const float*)d_in[1];
  const float* ln1w  = (const float*)d_in[2];
  const float* ln1b  = (const float*)d_in[3];
  const float* ln2w  = (const float*)d_in[4];
  const float* ln2b  = (const float*)d_in[5];
  const float* qkvLw = (const float*)d_in[6];
  const float* qkvLb = (const float*)d_in[7];
  const float* qkvRw = (const float*)d_in[8];
  const float* qkvRb = (const float*)d_in[9];
  const float* dwLw  = (const float*)d_in[10];
  const float* dwLb  = (const float*)d_in[11];
  const float* dwRw  = (const float*)d_in[12];
  const float* dwRb  = (const float*)d_in[13];
  const float* dil0w = (const float*)d_in[14];
  const float* dil0b = (const float*)d_in[15];
  const float* dil1w = (const float*)d_in[16];
  const float* dil1b = (const float*)d_in[17];
  const float* dil2w = (const float*)d_in[18];
  const float* dil2b = (const float*)d_in[19];
  const float* qw    = (const float*)d_in[20];
  const float* qb    = (const float*)d_in[21];
  const float* c1w   = (const float*)d_in[22];
  const float* c1b   = (const float*)d_in[23];
  const float* c2w   = (const float*)d_in[24];
  const float* c2b   = (const float*)d_in[25];
  const float* temp  = (const float*)d_in[26];

  char* ws = (char*)d_ws;
  size_t off = 0;
  auto alloc = [&](size_t bytes) -> void* {
    void* p = ws + off;
    off = (off + bytes + 255) & ~(size_t)255;
    return p;
  };
  half_t* normL = (half_t*)alloc((size_t)CBP * 48 * 2);
  half_t* normR = (half_t*)alloc((size_t)CBP * 48 * 2);
  half_t* wqkvL = (half_t*)alloc(144 * 64 * 2);
  half_t* wqkvR = (half_t*)alloc(144 * 64 * 2);
  half_t* wdil  = (half_t*)alloc(3 * 96 * 864 * 2);
  half_t* wqh   = (half_t*)alloc(48 * 288 * 2);
  half_t* qkvL  = (half_t*)alloc((size_t)CBP * 144 * 2);   // pixel-major [B][P][144]
  half_t* qkvR  = (half_t*)alloc((size_t)CBP * 144 * 2);
  half_t* Qcat  = (half_t*)alloc((size_t)CBP * 96 * 2);    // pixel-major [B][P][96]
  float*  KLb   = (float*)alloc((size_t)CBP * 48 * 4);     // channel-major
  float*  KRb   = (float*)alloc((size_t)CBP * 48 * 4);
  half_t* VLb   = (half_t*)alloc((size_t)CBP * 48 * 2);    // pixel-major [B][P][48]
  half_t* VRb   = (half_t*)alloc((size_t)CBP * 48 * 2);
  half_t* dcat  = (half_t*)alloc((size_t)CBP * 288 * 2);   // pixel-major [B][P][288]
  float*  Qfin  = (float*)alloc((size_t)CBP * 48 * 4);     // channel-major
  float*  invQ  = (float*)alloc(CB * 48 * 4);
  float*  invKL = (float*)alloc(CB * 48 * 4);
  float*  invKR = (float*)alloc(CB * 48 * 4);
  float*  Aat   = (float*)alloc(2 * CB * 3 * 256 * 4);
  half_t* Wc    = (half_t*)alloc(2 * CB * 48 * 64 * 2);
  (void)alloc(4096);  // overread slack for tail B-fragments

  k_prepack<<<1098, 256, 0, stream>>>(qkvLw, qkvRw, dil0w, dil1w, dil2w, qw,
                                      wqkvL, wqkvR, wdil, wqh);
  k_ln<<<512, 256, 0, stream>>>(x_l, x_r, ln1w, ln1b, ln2w, ln2b, normL, normR);
  k_qkv<<<2304, 256, 0, stream>>>(normL, normR, wqkvL, wqkvR, qkvLb, qkvRb,
                                  qkvL, qkvR);
  k_dw<<<147456, 256, 0, stream>>>(qkvL, qkvR, dwLw, dwLb, dwRw, dwRb, Qcat,
                                   KLb, KRb, VLb, VRb);
  k_dil<<<2304, 256, 0, stream>>>(Qcat, wdil, dil0b, dil1b, dil2b, dcat);
  k_convq<<<384, 256, 0, stream>>>(dcat, wqh, qb, Qfin);
  k_l2inv<<<288, 256, 0, stream>>>(Qfin, KLb, KRb, invQ, invKL, invKR);
  k_attn<<<12, 256, 0, stream>>>(Qfin, KLb, KRb, invQ, invKL, invKR, temp, Aat);
  k_wc<<<48, 256, 0, stream>>>(Aat, c1w, c2w, Wc);
  k_out<<<768, 256, 0, stream>>>(VLb, VRb, Wc, x_l, x_r, c1b, c2b, (float*)d_out);
}
